// AuxPEAttention_30537217474913
// MI455X (gfx1250) — matrix operand order repaired and checked by reading, not compiled, not run on hardware
//
#include <hip/hip_runtime.h>
#include <stdint.h>

typedef __bf16 bf16_t;
typedef __attribute__((ext_vector_type(8)))  __bf16 v8bf;
typedef __attribute__((ext_vector_type(16))) __bf16 v16bf;
typedef __attribute__((ext_vector_type(8)))  float  v8f;

#define B_  2
#define T_  2048
#define C_  1024
#define A_  256
#define CA_ 1280
#define H_  16
#define HD_ 64
#define M_  (B_ * T_)

__device__ __forceinline__ v8f wmma_bf16(v16bf a, v16bf b, v8f c) {
  return __builtin_amdgcn_wmma_f32_16x16x32_bf16(false, a, false, b, (short)0, c,
                                                 false, false);
}

// A fragment: 16x32 bf16 tile at (row0,k0) of row-major [*,ldk] buffer.
__device__ __forceinline__ v16bf ld_afrag(const bf16_t* base, int row0, int k0,
                                          int ldk, int lane) {
  int r = lane & 15, g = lane >> 4;
  const bf16_t* p = base + (size_t)(row0 + r) * ldk + k0 + g * 8;
  union { struct { v8bf a, b; } s; v16bf v; } u;
  u.s.a = *(const v8bf*)(p);
  u.s.b = *(const v8bf*)(p + 16);
  return u.v;
}

// B fragment: 32x16 (KxN) tile; source is B^T stored row-major [N,ldk].
__device__ __forceinline__ v16bf ld_bfrag(const bf16_t* base, int n0, int k0,
                                          int ldk, int lane) {
  int n = lane & 15, g = lane >> 4;
  const bf16_t* p = base + (size_t)(n0 + n) * ldk + k0 + g * 8;
  union { struct { v8bf a, b; } s; v16bf v; } u;
  u.s.a = *(const v8bf*)(p);
  u.s.b = *(const v8bf*)(p + 16);
  return u.v;
}

// ---------------- pack [x | aux] into bf16 x_aug [M_, CA_] ------------------
__global__ __launch_bounds__(256) void build_xaug(const float* __restrict__ x,
                                                  const float* __restrict__ aux,
                                                  bf16_t* __restrict__ xaug) {
  uint32_t i = blockIdx.x * 256u + threadIdx.x;   // exactly M_*CA_ threads
  uint32_t m = i / CA_, c = i % CA_;
  float v = (c < C_) ? x[(size_t)m * C_ + c] : aux[(size_t)m * A_ + (c - C_)];
  xaug[i] = (bf16_t)v;
}

// ---------------- W [K][C_] fp32  ->  Wt [C_][K] bf16 -----------------------
__global__ __launch_bounds__(256) void transpose_w(const float* __restrict__ W,
                                                   bf16_t* __restrict__ Wt,
                                                   int K) {
  __shared__ float tile[32][33];
  int k0 = blockIdx.x * 32, n0 = blockIdx.y * 32;
  int tr = threadIdx.x >> 5, tc = threadIdx.x & 31;
#pragma unroll
  for (int i = 0; i < 4; ++i) {
    int kk = tr + i * 8;
    tile[kk][tc] = W[(size_t)(k0 + kk) * C_ + n0 + tc];
  }
  __syncthreads();
#pragma unroll
  for (int i = 0; i < 4; ++i) {
    int nn = tr + i * 8;
    Wt[(size_t)(n0 + nn) * K + k0 + tc] = (bf16_t)tile[tc][nn];
  }
}

// ---------------- 128x128 bf16 WMMA GEMM, k-tile 64, fused epilogue ---------
// out(m,n) = (A[m,:K] @ Wt[n,:K]^T + bias[n] (+ pe[fi[m],n])) * scale
template <bool PE, bool F32OUT>
__global__ __launch_bounds__(256) void gemm128(
    const bf16_t* __restrict__ Ag, int lda, int K,
    const bf16_t* __restrict__ Wt,    // [C_][K] bf16 (pre-transposed)
    const float* __restrict__ bias,   // [C_]
    const float* __restrict__ pe,     // [MAX_LEN][C_] or unused
    const int* __restrict__ fi,       // [M_]
    float scale,
    bf16_t* __restrict__ obf, float* __restrict__ of) {
  __shared__ __align__(16) bf16_t As [128][64];   // 16 KB
  __shared__ __align__(16) bf16_t Bts[128][64];   // 16 KB (B^T: [n][k])
  int t = threadIdx.x;
  int w = t >> 5, lane = t & 31;
  int wr = w & 3, wc = w >> 2;                    // 4x2 wave grid -> 32x64/wave
  int m0 = blockIdx.x * 128, n0 = blockIdx.y * 128;
  int dc = t & 31, r0 = t >> 5;
  v8f acc[2][4] = {};
  for (int kt = 0; kt < K; kt += 64) {
    __syncthreads();
#pragma unroll
    for (int i = 0; i < 16; ++i) {  // stage A tile (dword copies, coalesced)
      int row = r0 + i * 8;
      ((uint32_t*)&As[row][0])[dc] =
          ((const uint32_t*)(Ag + (size_t)(m0 + row) * lda + kt))[dc];
    }
#pragma unroll
    for (int i = 0; i < 16; ++i) {  // stage B^T tile (bf16 already transposed)
      int row = r0 + i * 8;
      ((uint32_t*)&Bts[row][0])[dc] =
          ((const uint32_t*)(Wt + (size_t)(n0 + row) * K + kt))[dc];
    }
    __syncthreads();
    if (kt + 64 < K) {  // prefetch next tiles (one 128B line per row)
      if (t < 128) __builtin_prefetch(Ag + (size_t)(m0 + t) * lda + kt + 64, 0, 3);
      else __builtin_prefetch(Wt + (size_t)(n0 + (t - 128)) * K + kt + 64, 0, 3);
    }
#pragma unroll
    for (int ks = 0; ks < 2; ++ks) {
      v16bf a0 = ld_afrag(&As[0][0], wr * 32,      ks * 32, 64, lane);
      v16bf a1 = ld_afrag(&As[0][0], wr * 32 + 16, ks * 32, 64, lane);
#pragma unroll
      for (int ct = 0; ct < 4; ++ct) {
        v16bf b = ld_bfrag(&Bts[0][0], wc * 64 + ct * 16, ks * 32, 64, lane);
        acc[0][ct] = wmma_bf16(a0, b, acc[0][ct]);
        acc[1][ct] = wmma_bf16(a1, b, acc[1][ct]);
      }
    }
  }
  int hi = lane >> 4, ln = lane & 15;
  float biasv[4];
#pragma unroll
  for (int ct = 0; ct < 4; ++ct) biasv[ct] = bias[n0 + wc * 64 + ct * 16 + ln];

  if (F32OUT) {  // fp32 flat: stores are already 256B-contiguous per row
#pragma unroll
    for (int rt = 0; rt < 2; ++rt)
#pragma unroll
      for (int v = 0; v < 8; ++v) {
        int gm = m0 + wr * 32 + rt * 16 + v + 8 * hi;
#pragma unroll
        for (int ct = 0; ct < 4; ++ct) {
          int gn = n0 + wc * 64 + ct * 16 + ln;
          of[(size_t)gm * C_ + gn] = (acc[rt][ct][v] + biasv[ct]) * scale;
        }
      }
  } else {
    // Stage the wave's 32x64 result in LDS, then emit 128B-contiguous rows.
    __syncthreads();  // all waves done reading As/Bts fragments
    bf16_t (*St)[64] = (wc == 0) ? As : Bts;  // wave-private rows [wr*32, +32)
#pragma unroll
    for (int rt = 0; rt < 2; ++rt)
#pragma unroll
      for (int v = 0; v < 8; ++v) {
        int lr = wr * 32 + rt * 16 + v + 8 * hi;
        int gm = m0 + lr;
        const float* per = PE ? (pe + (size_t)fi[gm] * C_) : nullptr;
#pragma unroll
        for (int ct = 0; ct < 4; ++ct) {
          float val = acc[rt][ct][v] + biasv[ct];
          if (PE) val += per[n0 + wc * 64 + ct * 16 + ln];
          St[lr][ct * 16 + ln] = (bf16_t)(val * scale);
        }
      }
    // one row (64 bf16 = 128B) per lane, head-major destination
    int lr = wr * 32 + lane;
    int gm = m0 + lr;
    int b = gm >> 11, tt = gm & (T_ - 1);
    int h0 = (n0 + wc * 64) >> 6;
    uint4* dst = (uint4*)(obf + ((size_t)(b * H_ + h0) * T_ + tt) * HD_);
    const uint4* src = (const uint4*)&St[lr][0];
#pragma unroll
    for (int i = 0; i < 8; ++i) dst[i] = src[i];
  }
}

// ---------------- flash attention (bf16 WMMA, online softmax) ---------------
__global__ __launch_bounds__(256) void attn128(
    const bf16_t* __restrict__ qbf, const bf16_t* __restrict__ kbf,
    const bf16_t* __restrict__ vbf, const unsigned char* __restrict__ mask,
    bf16_t* __restrict__ attb) {
  __shared__ __align__(16) bf16_t Qs[128][HD_];   // 16 KB
  __shared__ __align__(16) bf16_t Ks[64][HD_];    //  8 KB (== K^T's B^T layout)
  __shared__ __align__(16) bf16_t Vt[HD_][64];    //  8 KB (V transposed)
  __shared__ __align__(16) bf16_t Ps[8][16][64];  // 16 KB (wave-private P)
  __shared__ __align__(4) unsigned char Ms[128][64];  // 8 KB mask tile
  int bh = blockIdx.x;
  int bb = bh >> 4, h = bh & 15;
  int q0 = blockIdx.y * 128;
  int t = threadIdx.x, w = t >> 5, lane = t & 31;
  int hi = lane >> 4, ln = lane & 15;
  const bf16_t* qh = qbf + (size_t)bh * T_ * HD_;
  const bf16_t* kh = kbf + (size_t)bh * T_ * HD_;
  const bf16_t* vh = vbf + (size_t)bh * T_ * HD_;
  const unsigned char* mbase = mask + (size_t)bb * T_ * T_;
#pragma unroll
  for (int i = 0; i < 16; ++i) {  // stage Q tile once
    int idx = t + i * 256;
    int row = idx >> 5, dcq = idx & 31;
    ((uint32_t*)&Qs[row][0])[dcq] =
        ((const uint32_t*)(qh + (size_t)(q0 + row) * HD_))[dcq];
  }
  __syncthreads();
  int qr = w * 16;  // wave owns 16 query rows
  v16bf aq[2];      // hoisted Q fragments (Qs is immutable from here on)
#pragma unroll
  for (int ks = 0; ks < 2; ++ks)
    aq[ks] = ld_afrag(&Qs[0][0], qr, ks * 32, HD_, lane);

  v8f o[4] = {};
  float mrow[8], lrow[8];
#pragma unroll
  for (int v = 0; v < 8; ++v) { mrow[v] = -1e30f; lrow[v] = 0.f; }

  for (int kt = 0; kt < T_; kt += 64) {
    __syncthreads();
#pragma unroll
    for (int i = 0; i < 8; ++i) {  // stage K tile row-major
      int idx = t + i * 256;
      int row = idx >> 5, dck = idx & 31;
      ((uint32_t*)&Ks[row][0])[dck] =
          ((const uint32_t*)(kh + (size_t)(kt + row) * HD_))[dck];
    }
#pragma unroll
    for (int i = 0; i < 16; ++i) {  // stage V transposed
      int idx = t + i * 256;
      int key = idx >> 6, d = idx & 63;
      Vt[d][key] = vh[(size_t)(kt + key) * HD_ + d];
    }
#pragma unroll
    for (int i = 0; i < 8; ++i) {  // stage mask tile (dword copies)
      int idx = t + i * 256;
      int row = idx >> 4, dcm = idx & 15;
      ((uint32_t*)&Ms[row][0])[dcm] =
          ((const uint32_t*)(mbase + (size_t)(q0 + row) * T_ + kt))[dcm];
    }
    __syncthreads();
    if (kt + 64 < T_) {  // prefetch next K/V tiles (one line per row)
      if (t < 64) __builtin_prefetch(kh + (size_t)(kt + 64 + t) * HD_, 0, 3);
      else if (t < 128)
        __builtin_prefetch(vh + (size_t)(kt + t) * HD_, 0, 3);  // t-64+64
    }
    v8f s[4] = {};
#pragma unroll
    for (int ks = 0; ks < 2; ++ks) {  // S = Q K^T  (1/sqrt(hd) folded into q)
#pragma unroll
      for (int ct = 0; ct < 4; ++ct) {
        v16bf b = ld_bfrag(&Ks[0][0], ct * 16, ks * 32, HD_, lane);
        s[ct] = wmma_bf16(aq[ks], b, s[ct]);
      }
    }
    float alpha[8];
#pragma unroll
    for (int v = 0; v < 8; ++v) {  // mask + online softmax per owned row
      int lr = qr + v + 8 * hi;
      float mx = mrow[v];
#pragma unroll
      for (int ct = 0; ct < 4; ++ct) {
        float sv = Ms[lr][ct * 16 + ln] ? s[ct][v] : -1e30f;
        s[ct][v] = sv;
        mx = fmaxf(mx, sv);
      }
      mx = fmaxf(mx, __shfl_xor(mx, 1, 32));
      mx = fmaxf(mx, __shfl_xor(mx, 2, 32));
      mx = fmaxf(mx, __shfl_xor(mx, 4, 32));
      mx = fmaxf(mx, __shfl_xor(mx, 8, 32));
      alpha[v] = __expf(mrow[v] - mx);
      float rs = 0.f;
#pragma unroll
      for (int ct = 0; ct < 4; ++ct) {
        float sv = s[ct][v];
        float p = (sv <= -1e29f) ? 0.f : __expf(sv - mx);
        Ps[w][v + 8 * hi][ct * 16 + ln] = (bf16_t)p;
        rs += p;
      }
      rs += __shfl_xor(rs, 1, 32);
      rs += __shfl_xor(rs, 2, 32);
      rs += __shfl_xor(rs, 4, 32);
      rs += __shfl_xor(rs, 8, 32);
      lrow[v] = lrow[v] * alpha[v] + rs;
      mrow[v] = mx;
    }
#pragma unroll
    for (int ct = 0; ct < 4; ++ct)
#pragma unroll
      for (int v = 0; v < 8; ++v) o[ct][v] *= alpha[v];
#pragma unroll
    for (int ks = 0; ks < 2; ++ks) {  // O += P V
      v16bf a = ld_afrag(&Ps[w][0][0], 0, ks * 32, 64, lane);
#pragma unroll
      for (int ct = 0; ct < 4; ++ct) {
        v16bf b = ld_bfrag(&Vt[0][0], ct * 16, ks * 32, 64, lane);
        o[ct] = wmma_bf16(a, b, o[ct]);
      }
    }
  }
  // normalize (fully-masked rows -> 0), stage in wave-private Ps, store 128B rows
#pragma unroll
  for (int v = 0; v < 8; ++v) {
    float linv = lrow[v] > 0.f ? 1.f / lrow[v] : 0.f;
#pragma unroll
    for (int ct = 0; ct < 4; ++ct)
      Ps[w][v + 8 * hi][ct * 16 + ln] = (bf16_t)(o[ct][v] * linv);
  }
  {
    int r = lane >> 1, half = lane & 1;  // 2 lanes per row, 64B each
    int gq = q0 + qr + r;
    uint4* dst = (uint4*)(attb + ((size_t)bb * T_ + gq) * C_ + h * HD_ + half * 32);
    const uint4* src = (const uint4*)&Ps[w][r][half * 32];
#pragma unroll
    for (int i = 0; i < 4; ++i) dst[i] = src[i];
  }
}

// ---------------- Launcher --------------------------------------------------
extern "C" void kernel_launch(void* const* d_in, const int* in_sizes, int n_in,
                              void* d_out, int out_size, void* d_ws,
                              size_t ws_size, hipStream_t stream) {
  const float* x   = (const float*)d_in[0];
  const float* aux = (const float*)d_in[1];
  const float* Wq  = (const float*)d_in[2];
  const float* bq  = (const float*)d_in[3];
  const float* Wk  = (const float*)d_in[4];
  const float* bk  = (const float*)d_in[5];
  const float* Wv  = (const float*)d_in[6];
  const float* bv  = (const float*)d_in[7];
  const float* Wo  = (const float*)d_in[8];
  const float* bo  = (const float*)d_in[9];
  const float* PEq = (const float*)d_in[10];
  const float* PEk = (const float*)d_in[11];
  const unsigned char* mask = (const unsigned char*)d_in[12];
  const int* fi = (const int*)d_in[13];
  float* out = (float*)d_out;

  char* ws = (char*)d_ws;
  bf16_t* xaug = (bf16_t*)(ws);              // 10,485,760 B
  bf16_t* qbf  = (bf16_t*)(ws + 10485760);   //  8,388,608 B
  bf16_t* kbf  = (bf16_t*)(ws + 18874368);   //  8,388,608 B
  bf16_t* vbf  = (bf16_t*)(ws + 27262976);   //  8,388,608 B
  bf16_t* attb = (bf16_t*)(ws + 35651584);   //  8,388,608 B
  bf16_t* wqT  = (bf16_t*)(ws + 44040192);   //  2,097,152 B  [C_][C_]
  bf16_t* wkT  = (bf16_t*)(ws + 46137344);   //  2,621,440 B  [C_][CA_]
  bf16_t* wvT  = (bf16_t*)(ws + 48758784);   //  2,621,440 B  [C_][CA_]
  bf16_t* woT  = (bf16_t*)(ws + 51380224);   //  2,097,152 B  (end ~53.5 MB)

  build_xaug<<<(M_ * CA_) / 256, 256, 0, stream>>>(x, aux, xaug);
  transpose_w<<<dim3(C_ / 32,  C_ / 32), 256, 0, stream>>>(Wq, wqT, C_);
  transpose_w<<<dim3(CA_ / 32, C_ / 32), 256, 0, stream>>>(Wk, wkT, CA_);
  transpose_w<<<dim3(CA_ / 32, C_ / 32), 256, 0, stream>>>(Wv, wvT, CA_);
  transpose_w<<<dim3(C_ / 32,  C_ / 32), 256, 0, stream>>>(Wo, woT, C_);

  dim3 g(M_ / 128, C_ / 128);  // 32 x 8
  // q = (x@Wq + bq + PEq[fi]) * 1/sqrt(64)  -> head-major bf16
  gemm128<true, false><<<g, 256, 0, stream>>>(xaug, CA_, C_, wqT, bq, PEq, fi,
                                              0.125f, qbf, nullptr);
  // k = x_aug@Wk + bk + PEk[fi]
  gemm128<true, false><<<g, 256, 0, stream>>>(xaug, CA_, CA_, wkT, bk, PEk, fi,
                                              1.0f, kbf, nullptr);
  // v = x_aug@Wv + bv
  gemm128<false, false><<<g, 256, 0, stream>>>(xaug, CA_, CA_, wvT, bv, nullptr,
                                               fi, 1.0f, vbf, nullptr);

  attn128<<<dim3(B_ * H_, T_ / 128), 256, 0, stream>>>(qbf, kbf, vbf, mask, attb);

  // out = att @ Wo + bo  (fp32 result)
  gemm128<false, true><<<g, 256, 0, stream>>>(attb, C_, C_, woT, bo, nullptr,
                                              fi, 1.0f, nullptr, out);
}
